// KPConvFPN_19739669692918
// MI455X (gfx1250) — matrix operand order repaired and checked by reading, not compiled, not run on hardware
//
#include <hip/hip_runtime.h>
#include <hip/hip_bf16.h>
#include <stdint.h>

#define NPTS  16384
#define CFEAT 128
#define CH2   64
#define KNN   20
#define BNEPS 1e-5f

typedef __attribute__((ext_vector_type(16))) __bf16 v16bf;
typedef __attribute__((ext_vector_type(8)))  float  v8f;
typedef __attribute__((ext_vector_type(4)))  unsigned int u32x4;
typedef __attribute__((ext_vector_type(8)))  int          i32x8;
typedef __attribute__((ext_vector_type(4)))  int          i32x4;

union FragBF { v16bf v; uint32_t w[8]; };

__device__ __forceinline__ uint16_t f2bf(float x) {
    uint32_t u = __float_as_uint(x);
    uint32_t r = (u + 0x7FFFu + ((u >> 16) & 1u)) >> 16;   // RNE
    return (uint16_t)r;
}
__device__ __forceinline__ uint32_t pack2bf(float a, float b) {
    return (uint32_t)f2bf(a) | ((uint32_t)f2bf(b) << 16);
}

// ---------------------------------------------------------------- xx = sum(xyz^2)
__global__ void k_xx(const float* __restrict__ xyz, float* __restrict__ xx) {
    int i = blockIdx.x * blockDim.x + threadIdx.x;
    if (i < NPTS) {
        float x = xyz[i], y = xyz[NPTS + i], z = xyz[2 * NPTS + i];
        xx[i] = x * x + y * y + z * z;
    }
}

// ---------------------------------------------------------------- KNN top-20 of pd
__global__ void k_knn(const float* __restrict__ xyz, const float* __restrict__ xx,
                      int* __restrict__ idx) {
    const int t = threadIdx.x;
    const int i = blockIdx.x * 256 + t;
    const float xi = xyz[i], yi = xyz[NPTS + i], zi = xyz[2 * NPTS + i], xxi = xx[i];
    float bv[KNN]; int bi[KNN];
#pragma unroll
    for (int s = 0; s < KNN; ++s) { bv[s] = -3.0e38f; bi[s] = 0; }
    __shared__ float sx[256], sy[256], sz[256], sq[256];
    for (int j0 = 0; j0 < NPTS; j0 += 256) {
        __syncthreads();
        sx[t] = xyz[j0 + t];
        sy[t] = xyz[NPTS + j0 + t];
        sz[t] = xyz[2 * NPTS + j0 + t];
        sq[t] = xx[j0 + t];
        __syncthreads();
        if (j0 + 256 < NPTS) __builtin_prefetch(&xyz[j0 + 256 + t], 0, 0);
#pragma unroll 4
        for (int jj = 0; jj < 256; ++jj) {
            float d = 2.0f * (xi * sx[jj] + yi * sy[jj] + zi * sz[jj]) - xxi - sq[jj];
            if (d > bv[KNN - 1]) {
                bv[KNN - 1] = d; bi[KNN - 1] = j0 + jj;
#pragma unroll
                for (int s = KNN - 1; s > 0; --s) {
                    if (bv[s] > bv[s - 1]) {
                        float tv = bv[s]; bv[s] = bv[s - 1]; bv[s - 1] = tv;
                        int   ti = bi[s]; bi[s] = bi[s - 1]; bi[s - 1] = ti;
                    }
                }
            }
        }
    }
#pragma unroll
    for (int s = 0; s < KNN; ++s) idx[i * KNN + s] = bi[s];
}

// ---------------------------------------------------------------- (C,N) -> (N,C)
__global__ void k_transpose(const float* __restrict__ F, float* __restrict__ Ft) {
    __shared__ float tile[32][33];
    int n0 = blockIdx.x * 32, c0 = blockIdx.y * 32;
    int tx = threadIdx.x & 31, ty = threadIdx.x >> 5;     // 32 x 8
#pragma unroll
    for (int i = 0; i < 32; i += 8)
        tile[ty + i][tx] = F[(c0 + ty + i) * NPTS + n0 + tx];
    __syncthreads();
#pragma unroll
    for (int i = 0; i < 32; i += 8)
        Ft[(size_t)(n0 + ty + i) * CFEAT + c0 + tx] = tile[tx][ty + i];
}

// ------------------------------------------- fold BN+bias, swizzle W2 to bf16 fragments
__global__ void k_prep(const float* __restrict__ b_mlp1,
                       const float* __restrict__ g1, const float* __restrict__ be1,
                       const float* __restrict__ m1, const float* __restrict__ v1,
                       const float* __restrict__ w_mlp2, const float* __restrict__ b_mlp2,
                       const float* __restrict__ g2, const float* __restrict__ be2,
                       const float* __restrict__ m2, const float* __restrict__ v2,
                       const float* __restrict__ b_up,
                       const float* __restrict__ gu, const float* __restrict__ beu,
                       const float* __restrict__ mu, const float* __restrict__ vu,
                       float* s1, float* t1, float* s2, float* t2,
                       float* su, float* tu, uint16_t* Wbf) {
    int t = threadIdx.x;
    if (t < CH2) {
        float sa = g1[t] * rsqrtf(v1[t] + BNEPS);
        s1[t] = sa; t1[t] = (b_mlp1[t] - m1[t]) * sa + be1[t];
        float sb = g2[t] * rsqrtf(v2[t] + BNEPS);
        s2[t] = sb; t2[t] = (b_mlp2[t] - m2[t]) * sb + be2[t];
    }
    if (t < CFEAT) {
        float sc = gu[t] * rsqrtf(vu[t] + BNEPS);
        su[t] = sc; tu[t] = (b_up[t] - mu[t]) * sc + beu[t];
    }
    // fragment-native B layout: frag fid = chunk*4+cb, lane holds 16 contiguous bf16
    for (int tid = t; tid < 8 * 4 * 512; tid += 256) {
        int fid = tid >> 9, rem = tid & 511, lane = rem >> 4, e = rem & 15;
        int i = e >> 1, half = e & 1;
        int klocal = 2 * i + half + ((lane >= 16) ? 16 : 0);
        int c = fid >> 2, cb = fid & 3;
        int k = c * 32 + klocal;
        int o = cb * 16 + (lane & 15);
        Wbf[tid] = f2bf(w_mlp2[o * 256 + k]);
    }
}

// ------------------------------- fused edge MLPs + relu/bn + max-pool over K (WMMA bf16)
struct EdgeSmem {
    uint16_t wb[8 * 4 * 512];        // 32 KB: full swizzled W2, TDM-staged
    uint16_t sA[8][32 * 32];         // 16 KB: per-wave A tiles
    int      idxs[8][KNN];
    float    nx[8][3][KNN];
    float    cf[8][32];
};

__global__ void __launch_bounds__(256) k_edge(
    const float* __restrict__ Ft, const float* __restrict__ xyz,
    const int* __restrict__ knn, const uint16_t* __restrict__ Wbf,
    const float* __restrict__ w1,
    const float* __restrict__ s1, const float* __restrict__ t1,
    const float* __restrict__ s2, const float* __restrict__ t2,
    float* __restrict__ fout) {
    const int wid  = threadIdx.x >> 5;
    const int lane = threadIdx.x & 31;
    const int n    = blockIdx.x * 8 + wid;     // one wave per point

    __shared__ EdgeSmem smem;

    // ---- stage the 32KB weight block into LDS once per workgroup ----
#if defined(__HIP_DEVICE_COMPILE__) && __has_builtin(__builtin_amdgcn_tensor_load_to_lds)
    if (wid == 0) {
        uint64_t ga = (uint64_t)(uintptr_t)Wbf;
        u32x4 g0;
        g0[0] = 1u;                                   // count=1, user mode, no gather
        g0[1] = (unsigned)offsetof(EdgeSmem, wb);     // lds_addr
        g0[2] = (unsigned)(ga & 0xFFFFFFFFu);         // global_addr[31:0]
        g0[3] = (unsigned)((ga >> 32) & 0x1FFFFFFu) | (2u << 30);  // addr[56:32], type=2
        i32x8 g1;
        g1[0] = (int)(3u << 16);                      // workgroup_mask=0, data_size=8B
        g1[1] = (int)(4096u << 16);                   // tensor_dim0 = 4096 (8B units)
        g1[2] = (int)(1u << 16);                      // tensor_dim0 hi=0, tensor_dim1=1
        g1[3] = (int)(4096u << 16);                   // tensor_dim1 hi=0, tile_dim0=4096
        g1[4] = 1;                                    // tile_dim1=1, tile_dim2=0
        g1[5] = 4096;                                 // tensor_dim0_stride lo
        g1[6] = (int)(4096u << 16);                   // stride0 hi=0, tensor_dim1_stride lo
        g1[7] = 0;
        i32x4 gz = {0, 0, 0, 0};
#if __clang_major__ >= 23
        i32x8 gz8 = {0, 0, 0, 0, 0, 0, 0, 0};
        __builtin_amdgcn_tensor_load_to_lds(g0, g1, gz, gz, gz8, 0);
#else
        __builtin_amdgcn_tensor_load_to_lds(g0, g1, gz, gz, 0);
#endif
#if __has_builtin(__builtin_amdgcn_s_wait_tensorcnt)
        __builtin_amdgcn_s_wait_tensorcnt(0);
#else
        asm volatile("s_wait_tensorcnt 0x0" ::: "memory");
#endif
    }
#else
    for (int tt = threadIdx.x; tt < (8 * 4 * 512) / 8; tt += 256)
        ((uint4*)smem.wb)[tt] = ((const uint4*)Wbf)[tt];
#endif

    if (lane < KNN) {
        int j = knn[n * KNN + lane];
        smem.idxs[wid][lane]  = j;
        smem.nx[wid][0][lane] = xyz[j];
        smem.nx[wid][1][lane] = xyz[NPTS + j];
        smem.nx[wid][2][lane] = xyz[2 * NPTS + j];
    }
    // zero-pad A rows >= KNN once (they stay zero across all chunks)
    if (lane >= KNN) {
        uint32_t* zp = (uint32_t*)&smem.sA[wid][lane * 32];
#pragma unroll
        for (int q = 0; q < 16; ++q) zp[q] = 0u;
    }
    const float cx = xyz[n], cy = xyz[NPTS + n], cz = xyz[2 * NPTS + n];
    __syncthreads();                               // weights + own-wave staging visible

    v8f acc[2][4];
#pragma unroll
    for (int a = 0; a < 2; ++a)
#pragma unroll
        for (int b = 0; b < 4; ++b)
            acc[a][b] = (v8f){0.f, 0.f, 0.f, 0.f, 0.f, 0.f, 0.f, 0.f};

#pragma unroll
    for (int chi = 0; chi < 8; ++chi) {
        const int k0 = chi * 32;
        const int csrc = (k0 < CFEAT) ? k0 : (k0 - CFEAT);
        smem.cf[wid][lane] = Ft[(size_t)n * CFEAT + csrc + lane];   // coalesced row read
        asm volatile("s_wait_dscnt 0x0" ::: "memory");
        // stage A tile: lane = row, contiguous float4 reads from transposed features
        const int r = lane;
        if (r < KNN) {
            uint32_t* dst = (uint32_t*)&smem.sA[wid][r * 32];
            if (k0 < CFEAT) {                       // nf[:,0:128] = nbr - center
                int j = smem.idxs[wid][r];
                const float4* src = (const float4*)(Ft + (size_t)j * CFEAT + k0);
                const float4* cfp = (const float4*)&smem.cf[wid][0];
#pragma unroll
                for (int q = 0; q < 8; ++q) {
                    float4 v = src[q]; float4 c = cfp[q];
                    dst[2 * q]     = pack2bf(v.x - c.x, v.y - c.y);
                    dst[2 * q + 1] = pack2bf(v.z - c.z, v.w - c.w);
                }
            } else {                                // nf[:,128:256] = center (all rows)
                const float4* cfp = (const float4*)&smem.cf[wid][0];
#pragma unroll
                for (int q = 0; q < 8; ++q) {
                    float4 c = cfp[q];
                    dst[2 * q]     = pack2bf(c.x, c.y);
                    dst[2 * q + 1] = pack2bf(c.z, c.w);
                }
            }
        }
        asm volatile("s_wait_dscnt 0x0" ::: "memory");

        // A fragments, contiguous k order (k = 16*(lane/16) + element), matching B
        FragBF a0, a1;
        const int M   = lane & 15;
        const int kup = (lane >= 16) ? 16 : 0;
#pragma unroll
        for (int i = 0; i < 8; ++i) {
            int K = 2 * i + kup;
            a0.w[i] = *reinterpret_cast<const uint32_t*>(&smem.sA[wid][M * 32 + K]);
            a1.w[i] = *reinterpret_cast<const uint32_t*>(&smem.sA[wid][(M + 16) * 32 + K]);
        }
#pragma unroll
        for (int cb = 0; cb < 4; ++cb) {
            FragBF b;
            const uint32_t* wp =
                (const uint32_t*)&smem.wb[(chi * 4 + cb) * 512 + lane * 16];
#pragma unroll
            for (int i = 0; i < 8; ++i) b.w[i] = wp[i];
            acc[0][cb] = __builtin_amdgcn_wmma_f32_16x16x32_bf16(
                false, a0.v, false, b.v, (short)0, acc[0][cb], false, false);
            acc[1][cb] = __builtin_amdgcn_wmma_f32_16x16x32_bf16(
                false, a1.v, false, b.v, (short)0, acc[1][cb], false, false);
        }
    }

    // hf: bn(scale/shift) + relu + max over valid rows (tile0: 16, tile1: rows 16..19)
#pragma unroll
    for (int cb = 0; cb < 4; ++cb) {
        int o = cb * 16 + (lane & 15);
        float s = s2[o], tt = t2[o];
        float m = -3.0e38f;
#pragma unroll
        for (int rr = 0; rr < 8; ++rr)
            m = fmaxf(m, fmaxf(acc[0][cb][rr] * s + tt, 0.0f));
        if (lane < 16) {
#pragma unroll
            for (int rr = 0; rr < 4; ++rr)     // global rows 16..19 only
                m = fmaxf(m, fmaxf(acc[1][cb][rr] * s + tt, 0.0f));
        }
        float other = __shfl_xor(m, 16, 32);   // wave32: halves hold M and M+8
        m = fmaxf(m, other);
        if (lane < 16) fout[(size_t)n * CFEAT + CH2 + o] = m;
    }

    // hx: 6-dim MLP, each lane covers 2 of 64 output channels
#pragma unroll
    for (int h2 = 0; h2 < 2; ++h2) {
        int o = lane + 32 * h2;
        float w0 = w1[o * 6 + 0], wa = w1[o * 6 + 1], w2c = w1[o * 6 + 2];
        float w3 = w1[o * 6 + 3], w4 = w1[o * 6 + 4], w5  = w1[o * 6 + 5];
        float s = s1[o], tt = t1[o];
        float base = w3 * cx + w4 * cy + w5 * cz;
        float m = -3.0e38f;
        for (int rr = 0; rr < KNN; ++rr) {
            float dx = smem.nx[wid][0][rr] - cx;
            float dy = smem.nx[wid][1][rr] - cy;
            float dz = smem.nx[wid][2][rr] - cz;
            float z = w0 * dx + wa * dy + w2c * dz + base;
            m = fmaxf(m, fmaxf(s * z + tt, 0.0f));
        }
        fout[(size_t)n * CFEAT + o] = m;
    }
}

// ---------------------------------------------------------------- f1/f2 heads
__global__ void k_f1f2(const float* __restrict__ f, const float* __restrict__ wd1,
                       const float* __restrict__ wd2,
                       float* __restrict__ f1, float* __restrict__ f2) {
    int tid = blockIdx.x * blockDim.x + threadIdx.x;
    int n = tid >> 4, o = tid & 15;
    if (n >= NPTS) return;
    float a1 = 0.f, a2 = 0.f;
    const float* fr  = f + (size_t)n * CFEAT;
    const float* w1p = wd1 + o * CFEAT;
    const float* w2p = wd2 + o * CFEAT;
#pragma unroll 8
    for (int c = 0; c < CFEAT; ++c) { float fv = fr[c]; a1 += fv * w1p[c]; a2 += fv * w2p[c]; }
    f1[tid] = fmaxf(a1, 0.f);
    f2[tid] = fmaxf(a2, 0.f);
}

// ---------------------------------------------------------------- ch = mean_n f1
__global__ void k_ch(const float* __restrict__ f1, float* __restrict__ ch) {
    __shared__ float red[256];
    int o = blockIdx.x;
    float s = 0.f;
    for (int n = threadIdx.x; n < NPTS; n += 256) s += f1[n * 16 + o];
    red[threadIdx.x] = s; __syncthreads();
    for (int st = 128; st > 0; st >>= 1) {
        if (threadIdx.x < st) red[threadIdx.x] += red[threadIdx.x + st];
        __syncthreads();
    }
    if (threadIdx.x == 0) ch[o] = red[0] * (1.0f / NPTS);
}

// ---------------------------------------------------------------- sp = mean_c f2
__global__ void k_sp(const float* __restrict__ f2, float* __restrict__ sp) {
    int n = blockIdx.x * blockDim.x + threadIdx.x;
    if (n >= NPTS) return;
    float s = 0.f;
#pragma unroll
    for (int o = 0; o < 16; ++o) s += f2[n * 16 + o];
    sp[n] = s * (1.0f / 16.0f);
}

// ---------------------------------------------------------------- fin = fe + f1 + f2
__global__ void k_fin(const float* __restrict__ f1, const float* __restrict__ f2,
                      const float* __restrict__ ch, const float* __restrict__ sp,
                      float* __restrict__ fin) {
    int tid = blockIdx.x * blockDim.x + threadIdx.x;
    int n = tid >> 4, o = tid & 15;
    if (n >= NPTS) return;
    fin[tid] = sqrtf(ch[o] * sp[n] + 1e-12f) + f1[tid] + f2[tid];
}

// ----------------------------------------------- up-proj + mish + (C,N) transpose out
__global__ void k_final(const float* __restrict__ f, const float* __restrict__ fin,
                        const float* __restrict__ wup, const float* __restrict__ su,
                        const float* __restrict__ tu, float* __restrict__ out) {
    int tid = blockIdx.x * blockDim.x + threadIdx.x;
    int c = tid >> 14, n = tid & (NPTS - 1);      // out index == c*N + n == tid
    float a = 0.f;
    const float* fr = fin + n * 16;
    const float* wr = wup + c * 16;
#pragma unroll
    for (int o = 0; o < 16; ++o) a += fr[o] * wr[o];
    float up = fmaxf(su[c] * a + tu[c], 0.f);
    float x = f[(size_t)n * CFEAT + c] - up;
    float spv = (x > 20.f) ? x : log1pf(__expf(x));
    out[tid] = x * tanhf(spv);
}

// ================================================================ launch
extern "C" void kernel_launch(void* const* d_in, const int* in_sizes, int n_in,
                              void* d_out, int out_size, void* d_ws, size_t ws_size,
                              hipStream_t stream) {
    (void)in_sizes; (void)n_in; (void)out_size; (void)ws_size;

    const float* xyz    = (const float*)d_in[0];
    const float* feat   = (const float*)d_in[1];
    const float* w_mlp1 = (const float*)d_in[2];
    const float* b_mlp1 = (const float*)d_in[3];
    const float* bn1_g  = (const float*)d_in[4];
    const float* bn1_b  = (const float*)d_in[5];
    const float* bn1_m  = (const float*)d_in[6];
    const float* bn1_v  = (const float*)d_in[7];
    const float* w_mlp2 = (const float*)d_in[8];
    const float* b_mlp2 = (const float*)d_in[9];
    const float* bn2_g  = (const float*)d_in[10];
    const float* bn2_b  = (const float*)d_in[11];
    const float* bn2_m  = (const float*)d_in[12];
    const float* bn2_v  = (const float*)d_in[13];
    const float* w_d1   = (const float*)d_in[14];
    const float* w_d2   = (const float*)d_in[15];
    const float* w_up   = (const float*)d_in[16];
    const float* b_up   = (const float*)d_in[17];
    const float* bnu_g  = (const float*)d_in[18];
    const float* bnu_b  = (const float*)d_in[19];
    const float* bnu_m  = (const float*)d_in[20];
    const float* bnu_v  = (const float*)d_in[21];
    float* out = (float*)d_out;

    size_t off = 0;
    auto carve = [&](size_t bytes) -> void* {
        void* p = (char*)d_ws + off;
        off = (off + bytes + 255) & ~(size_t)255;
        return p;
    };
    float*    ws_xx  = (float*)carve(NPTS * 4);
    int*      ws_idx = (int*)carve((size_t)NPTS * KNN * 4);
    float*    ws_Ft  = (float*)carve((size_t)NPTS * CFEAT * 4);
    float*    ws_f   = (float*)carve((size_t)NPTS * CFEAT * 4);
    float*    ws_f1  = (float*)carve((size_t)NPTS * 16 * 4);
    float*    ws_f2  = (float*)carve((size_t)NPTS * 16 * 4);
    float*    ws_fin = (float*)carve((size_t)NPTS * 16 * 4);
    float*    ws_ch  = (float*)carve(16 * 4);
    float*    ws_sp  = (float*)carve(NPTS * 4);
    float*    ws_s1  = (float*)carve(CH2 * 4);
    float*    ws_t1  = (float*)carve(CH2 * 4);
    float*    ws_s2  = (float*)carve(CH2 * 4);
    float*    ws_t2  = (float*)carve(CH2 * 4);
    float*    ws_su  = (float*)carve(CFEAT * 4);
    float*    ws_tu  = (float*)carve(CFEAT * 4);
    uint16_t* ws_Wbf = (uint16_t*)carve(8 * 4 * 512 * 2);

    k_xx<<<NPTS / 256, 256, 0, stream>>>(xyz, ws_xx);
    k_knn<<<NPTS / 256, 256, 0, stream>>>(xyz, ws_xx, ws_idx);
    k_transpose<<<dim3(NPTS / 32, CFEAT / 32), 256, 0, stream>>>(feat, ws_Ft);
    k_prep<<<1, 256, 0, stream>>>(b_mlp1, bn1_g, bn1_b, bn1_m, bn1_v,
                                  w_mlp2, b_mlp2, bn2_g, bn2_b, bn2_m, bn2_v,
                                  b_up, bnu_g, bnu_b, bnu_m, bnu_v,
                                  ws_s1, ws_t1, ws_s2, ws_t2, ws_su, ws_tu, ws_Wbf);
    k_edge<<<NPTS / 8, 256, 0, stream>>>(ws_Ft, xyz, ws_idx, ws_Wbf, w_mlp1,
                                         ws_s1, ws_t1, ws_s2, ws_t2, ws_f);
    k_f1f2<<<(NPTS * 16) / 256, 256, 0, stream>>>(ws_f, w_d1, w_d2, ws_f1, ws_f2);
    k_ch<<<16, 256, 0, stream>>>(ws_f1, ws_ch);
    k_sp<<<NPTS / 256, 256, 0, stream>>>(ws_f2, ws_sp);
    k_fin<<<(NPTS * 16) / 256, 256, 0, stream>>>(ws_f1, ws_f2, ws_ch, ws_sp, ws_fin);
    k_final<<<(NPTS * CFEAT) / 256, 256, 0, stream>>>(ws_f, ws_fin, w_up, ws_su, ws_tu, out);
}
